// GAEEmbeddingModel_74852690034968
// MI455X (gfx1250) — matrix operand order repaired and checked by reading, not compiled, not run on hardware
//
#include <hip/hip_runtime.h>
#include <math.h>

#define NB 200000
#define EB 3200000
#define NP 100000
#define EP 1600000
#define CHUNK 50000

typedef __attribute__((ext_vector_type(16))) __bf16 v16bf;
typedef __attribute__((ext_vector_type(8)))  __bf16 v8bf;
typedef __attribute__((ext_vector_type(8)))  float  v8f;

__device__ __forceinline__ __bf16 f2bf(float f) {
  union { float f; unsigned u; } v; v.f = f;
  unsigned r = v.u + 0x7FFFu + ((v.u >> 16) & 1u);   // round-to-nearest-even
  unsigned short h = (unsigned short)(r >> 16);
  __bf16 o; __builtin_memcpy(&o, &h, sizeof(o));
  return o;
}

__device__ __forceinline__ v16bf combine(v8bf lo, v8bf hi) {
  return __builtin_shufflevector(lo, hi, 0,1,2,3,4,5,6,7,8,9,10,11,12,13,14,15);
}

// ---- staging: A (optionally concat of two sources) -> bf16, K padded to Kp with zeros
__global__ void k_cvt_a(const float* __restrict__ A, const float* __restrict__ A2,
                        int KA, int K, int Kp, int M, __bf16* __restrict__ out) {
  long i = (long)blockIdx.x * blockDim.x + threadIdx.x;
  long total = (long)M * Kp;
  if (i < total) {
    int r = (int)(i / Kp);
    int k = (int)(i % Kp);
    float v = 0.0f;
    if (k < KA)      v = A[(long)r * KA + k];
    else if (k < K)  v = A2[(long)r * (K - KA) + (k - KA)];
    out[i] = f2bf(v);
  }
}

// ---- staging: W[K,N] -> Wt[N,Kp] bf16 (transposed, zero-padded K)
__global__ void k_cvt_wt(const float* __restrict__ W, int K, int N, int Kp,
                         __bf16* __restrict__ out) {
  long i = (long)blockIdx.x * blockDim.x + threadIdx.x;
  long total = (long)N * Kp;
  if (i < total) {
    int n = (int)(i / Kp);
    int k = (int)(i % Kp);
    out[i] = f2bf((k < K) ? W[(long)k * N + n] : 0.0f);
  }
}

// Block-cooperative WMMA GEMM:
//   block = 8 waves; all waves share one group of NT 16-wide N tiles; wave w
//   handles M tile (blockM*8 + w). B chunk (NT*16 cols x 32 K) staged in LDS,
//   double-buffered, one barrier per K-step. Next-step A/B global loads are
//   issued before the WMMAs; all NT B fragments are pulled from LDS into
//   registers before the WMMA group so the MACs run back-to-back.
template <int NT>
__global__ __launch_bounds__(256) void k_gemm_wmma(
    const __bf16* __restrict__ Abf, const __bf16* __restrict__ Wt,
    const float* __restrict__ bias, const float* __restrict__ alpha,
    float* __restrict__ C, int M, int N, int Kp, int act) {
  constexpr int COLS = NT * 16;
  constexpr int RS   = 40;                 // LDS row stride in bf16 (32 + 8 pad)
  constexpr bool ALLSTAGE = (COLS * 4 == 256);
  __shared__ __bf16 lds[2][COLS][RS];

  const int groupsN = (N >> 4) / NT;
  const int tilesM  = M >> 4;
  const int mb = (int)blockIdx.x / groupsN;
  const int tg = (int)blockIdx.x % groupsN;
  const int tid  = (int)threadIdx.x;
  const int wave = tid >> 5;
  const int lane = tid & 31;
  const int tm   = mb * 8 + wave;
  const bool valid = (tm < tilesM);        // wave-uniform
  const int tmc  = valid ? tm : (tilesM - 1);
  const int m0   = tmc << 4;
  const int n0   = (tg * NT) << 4;
  const int hx   = lane >> 4;
  const int l15  = lane & 15;

  // B staging role for this thread (compile-time all-threads for NT==4)
  const int scol = tid >> 2;
  const int sseg = tid & 3;
  const bool stager = ALLSTAGE || (tid < COLS * 4);
  const __bf16* __restrict__ wrow = Wt + (long)(n0 + (scol < COLS ? scol : 0)) * Kp + sseg * 8;

  const __bf16* __restrict__ arow = Abf + (long)(m0 + l15) * Kp;

  v8f acc[NT];
#pragma unroll
  for (int t = 0; t < NT; ++t) acc[t] = (v8f){0.f,0.f,0.f,0.f,0.f,0.f,0.f,0.f};

  // prologue: stage k0=0 B chunk, prefetch k0=0 A fragment
  if (stager) *(v8bf*)&lds[0][scol][sseg * 8] = *(const v8bf*)(wrow + 0);
  v8bf a0 = *(const v8bf*)(arow + hx * 8);
  v8bf a1 = *(const v8bf*)(arow + 16 + hx * 8);
  __syncthreads();

  int buf = 0;
  for (int k0 = 0; k0 < Kp; k0 += 32) {
    const int kn = k0 + 32;
    const bool hasNext = (kn < Kp);        // uniform
    v8bf na0, na1, nb;
    if (hasNext) {                          // issue next-step global loads first
      na0 = *(const v8bf*)(arow + kn + hx * 8);
      na1 = *(const v8bf*)(arow + kn + 16 + hx * 8);
      if (stager) nb = *(const v8bf*)(wrow + kn);
    }
    // A 16x32 layout: lanes0-15 K={0..7,16..23}, lanes16-31 K={8..15,24..31}
    v16bf a = combine(a0, a1);
    // gather all B fragments from LDS first, then issue WMMAs back-to-back
    v16bf bfrag[NT];
#pragma unroll
    for (int t = 0; t < NT; ++t) {
      // B 32x16 layout: lanes0-15 K={0..7,16..23}, lanes16-31 K={8..15,24..31} (col = lane&15)
      const __bf16* bp = &lds[buf][t * 16 + l15][hx * 8];
      bfrag[t] = combine(*(const v8bf*)bp, *(const v8bf*)(bp + 16));
    }
#pragma unroll
    for (int t = 0; t < NT; ++t)
      acc[t] = __builtin_amdgcn_wmma_f32_16x16x32_bf16(false, a, false, bfrag[t],
                                                       (short)0, acc[t], false, false);
    if (hasNext && stager) *(v8bf*)&lds[buf ^ 1][scol][sseg * 8] = nb;
    __syncthreads();
    a0 = na0; a1 = na1; buf ^= 1;
  }

  if (valid) {                              // wave-uniform store guard
    float al = (act == 1) ? alpha[0] : 0.0f;
#pragma unroll
    for (int t = 0; t < NT; ++t) {
      int col = n0 + t * 16 + l15;
      float bs = (act == 1) ? bias[col] : 0.0f;
#pragma unroll
      for (int r = 0; r < 8; ++r) {
        long rr = m0 + r + hx * 8;          // C layout: VGPR r -> row r + 8*half
        float v = acc[t][r];
        if (act == 1) { v += bs; v = (v > 0.0f) ? v : al * v; }
        C[rr * (long)N + col] = v;
      }
    }
  }
}

__global__ void k_fill(float* __restrict__ p, float v, long n) {
  long i = (long)blockIdx.x * blockDim.x + threadIdx.x;
  if (i < n) p[i] = v;
}

__global__ void k_deg(const int* __restrict__ dst, int E, float* __restrict__ deg) {
  int i = blockIdx.x * blockDim.x + threadIdx.x;
  if (i < E)
    __hip_atomic_fetch_add(&deg[dst[i]], 1.0f, __ATOMIC_RELAXED, __HIP_MEMORY_SCOPE_AGENT);
}

__global__ void k_rsqrt(float* __restrict__ p, int n) {
  int i = blockIdx.x * blockDim.x + threadIdx.x;
  if (i < n) p[i] = rsqrtf(fmaxf(p[i], 1.0f));
}

// out[i,f] = dis[i]^2 * h[i,f] + bias[f]   (self-loop term + bias); F = 1<<Fs
__global__ void k_agg_init(const float* __restrict__ h, const float* __restrict__ dis,
                           const float* __restrict__ bias, float* __restrict__ out,
                           int n, int Fs) {
  long i = (long)blockIdx.x * blockDim.x + threadIdx.x;
  long total = (long)n << Fs;
  if (i < total) {
    int node = (int)(i >> Fs);
    int f = (int)(i & ((1 << Fs) - 1));
    float d = dis[node];
    out[i] = d * d * h[i] + bias[f];
  }
}

// out[dst] += dis[src]*dis[dst] * h[src]; 4 floats/thread; F = 1<<Fs
__global__ void k_agg_scatter(const float* __restrict__ h, const float* __restrict__ dis,
                              const int* __restrict__ src, const int* __restrict__ dst,
                              int E, int Fs, float* __restrict__ out) {
  long i = (long)blockIdx.x * blockDim.x + threadIdx.x;
  int cs = Fs - 2;
  long total = (long)E << cs;
  if (i < total) {
    int e = (int)(i >> cs);
    int c = (int)(i & ((1 << cs) - 1));
    int s = src[e], d = dst[e];
    float nm = dis[s] * dis[d];
    const float* hp = h + ((long)s << Fs) + c * 4;
    float* op = out + ((long)d << Fs) + c * 4;
#pragma unroll
    for (int j = 0; j < 4; ++j)
      __hip_atomic_fetch_add(op + j, nm * hp[j], __ATOMIC_RELAXED, __HIP_MEMORY_SCOPE_AGENT);
  }
}

__global__ void k_relu(float* __restrict__ p, long n) {
  long i = (long)blockIdx.x * blockDim.x + threadIdx.x;
  if (i < n) p[i] = fmaxf(p[i], 0.0f);
}

// sigmoid( dot32(z[src], z[dst]) )
__global__ void k_decode(const float* __restrict__ z, const int* __restrict__ src,
                         const int* __restrict__ dst, int E, float* __restrict__ out) {
  int e = blockIdx.x * blockDim.x + threadIdx.x;
  if (e < E) {
    const float4* a = (const float4*)(z + (long)src[e] * 32);
    const float4* b = (const float4*)(z + (long)dst[e] * 32);
    float dot = 0.0f;
#pragma unroll
    for (int j = 0; j < 8; ++j) {
      float4 x = a[j], y = b[j];
      dot += x.x * y.x + x.y * y.y + x.z * y.z + x.w * y.w;
    }
    out[e] = 1.0f / (1.0f + expf(-dot));
  }
}

static inline int cdiv(long a, long b) { return (int)((a + b - 1) / b); }

// full GEMM op: stage A->bf16 (padded, optional concat), stage W->Wt bf16, WMMA GEMM
static void gemm(const float* A, const float* A2, int KA, const float* W,
                 const float* bias, const float* alpha, float* C,
                 int M, int N, int K, int act, __bf16* Abf, __bf16* Wbf, hipStream_t s) {
  int Kp = (K + 31) & ~31;
  long ta = (long)M * Kp;
  k_cvt_a<<<cdiv(ta, 256), 256, 0, s>>>(A, A2 ? A2 : A, A2 ? KA : K, K, Kp, M, Abf);
  long tw = (long)N * Kp;
  k_cvt_wt<<<cdiv(tw, 256), 256, 0, s>>>(W, K, N, Kp, Wbf);
  int tilesM = M >> 4;
  if (((N >> 4) & 3) == 0) {
    int groupsN = (N >> 4) / 4;
    int blocks = cdiv(tilesM, 8) * groupsN;
    k_gemm_wmma<4><<<blocks, 256, 0, s>>>(Abf, Wbf, bias, alpha, C, M, N, Kp, act);
  } else {
    int groupsN = (N >> 4) / 2;
    int blocks = cdiv(tilesM, 8) * groupsN;
    k_gemm_wmma<2><<<blocks, 256, 0, s>>>(Abf, Wbf, bias, alpha, C, M, N, Kp, act);
  }
}

static void gcn_layer(const float* hmat, const float* dis, const float* bias,
                      const int* src, const int* dst, int E, int n, int Fs,
                      float* out, hipStream_t s) {
  long tot = (long)n << Fs;
  k_agg_init<<<cdiv(tot, 256), 256, 0, s>>>(hmat, dis, bias, out, n, Fs);
  long sc = (long)E << (Fs - 2);
  k_agg_scatter<<<cdiv(sc, 256), 256, 0, s>>>(hmat, dis, src, dst, E, Fs, out);
}

extern "C" void kernel_launch(void* const* d_in, const int* in_sizes, int n_in,
                              void* d_out, int out_size, void* d_ws, size_t ws_size,
                              hipStream_t stream) {
  (void)in_sizes; (void)n_in; (void)out_size; (void)ws_size;
  // ---- inputs (setup_inputs dict order) ----
  const float* in_bc = (const float*)d_in[0];   // [NB,110]
  const float* in_bd = (const float*)d_in[1];   // [NB,13]
  const float* in_pc = (const float*)d_in[2];   // [NP,401]
  const int*   eib   = (const int*)d_in[3];     // [2,EB]
  const int*   eip   = (const int*)d_in[4];     // [2,EP]
  const int *src_b = eib, *dst_b = eib + EB;
  const int *src_p = eip, *dst_p = eip + EP;
  const float *bc_w1=(const float*)d_in[5],  *bc_b1=(const float*)d_in[6],  *bc_a1=(const float*)d_in[7];
  const float *bc_w2=(const float*)d_in[8],  *bc_b2=(const float*)d_in[9],  *bc_a2=(const float*)d_in[10];
  const float *bd_w1=(const float*)d_in[11], *bd_b1=(const float*)d_in[12], *bd_a1=(const float*)d_in[13];
  const float *bd_w2=(const float*)d_in[14], *bd_b2=(const float*)d_in[15], *bd_a2=(const float*)d_in[16];
  const float *bf_w1=(const float*)d_in[17], *bf_b1=(const float*)d_in[18], *bf_a1=(const float*)d_in[19];
  const float *bf_w2=(const float*)d_in[20], *bf_b2=(const float*)d_in[21], *bf_a2=(const float*)d_in[22];
  const float *pc_w1=(const float*)d_in[23], *pc_b1=(const float*)d_in[24], *pc_a1=(const float*)d_in[25];
  const float *pc_w2=(const float*)d_in[26], *pc_b2=(const float*)d_in[27], *pc_a2=(const float*)d_in[28];
  const float *gb_w1=(const float*)d_in[29], *gb_b1=(const float*)d_in[30];
  const float *gb_w2=(const float*)d_in[31], *gb_b2=(const float*)d_in[32];
  const float *gp_w1=(const float*)d_in[33], *gp_b1=(const float*)d_in[34];
  const float *gp_w2=(const float*)d_in[35], *gp_b2=(const float*)d_in[36];

  // ---- outputs: z_building[NB*32] | adj_b[EB] | z_pois[NP*32] | adj_p[EP] ----
  float* out   = (float*)d_out;
  float* z_b   = out;
  float* adj_b = out + (long)NB * 32;
  float* z_p   = adj_b + EB;
  float* adj_p = z_p + (long)NP * 32;

  // ---- workspace layout (floats; all offsets multiples of 4 -> 16B aligned) ----
  float* ws     = (float*)d_ws;
  float* hidden = ws;                          // CHUNK*512 = 25,600,000 f
  float* cat_c  = hidden + (long)CHUNK * 512;  // CHUNK*64
  float* desc_c = cat_c  + (long)CHUNK * 64;   // CHUNK*32
  float* tmp64  = desc_c + (long)CHUNK * 32;   // CHUNK*64
  float* xb     = tmp64  + (long)CHUNK * 64;   // NB*64
  float* hw     = xb     + (long)NB * 64;      // NB*64
  float* out1   = hw     + (long)NB * 64;      // NB*64
  float* hw2    = out1   + (long)NB * 64;      // NB*32
  float* deg    = hw2    + (long)NB * 32;      // NB
  __bf16* Abf   = (__bf16*)(deg + NB);         // max(CHUNK*512, NB*64) = 25.6M bf16
  __bf16* Wbf   = Abf + (long)CHUNK * 512;     // max 512*512 bf16

  // ============ buildings: MLP stacks (chunked over rows) ============
  for (int c = 0; c < NB / CHUNK; ++c) {
    long r0 = (long)c * CHUNK;
    gemm(in_bc + r0 * 110, nullptr, 110, bc_w1, bc_b1, bc_a1, hidden, CHUNK, 512, 110, 1, Abf, Wbf, stream);
    gemm(hidden, nullptr, 512, bc_w2, bc_b2, bc_a2, cat_c, CHUNK, 64, 512, 1, Abf, Wbf, stream);
    gemm(in_bd + r0 * 13, nullptr, 13, bd_w1, bd_b1, bd_a1, tmp64, CHUNK, 64, 13, 1, Abf, Wbf, stream);
    gemm(tmp64, nullptr, 64, bd_w2, bd_b2, bd_a2, desc_c, CHUNK, 32, 64, 1, Abf, Wbf, stream);
    // concat(cat,desc) folded into the A-staging kernel (KA=64, K=96)
    gemm(cat_c, desc_c, 64, bf_w1, bf_b1, bf_a1, hidden, CHUNK, 512, 96, 1, Abf, Wbf, stream);
    gemm(hidden, nullptr, 512, bf_w2, bf_b2, bf_a2, xb + r0 * 64, CHUNK, 64, 512, 1, Abf, Wbf, stream);
  }

  // ============ buildings: GCN encoder ============
  gemm(xb, nullptr, 64, gb_w1, nullptr, nullptr, hw, NB, 64, 64, 0, Abf, Wbf, stream);
  k_fill<<<cdiv(NB, 256), 256, 0, stream>>>(deg, 1.0f, NB);         // self loop
  k_deg<<<cdiv(EB, 256), 256, 0, stream>>>(dst_b, EB, deg);
  k_rsqrt<<<cdiv(NB, 256), 256, 0, stream>>>(deg, NB);              // deg -> dis
  gcn_layer(hw, deg, gb_b1, src_b, dst_b, EB, NB, 6, out1, stream); // F=64
  k_relu<<<cdiv((long)NB * 64, 256), 256, 0, stream>>>(out1, (long)NB * 64);
  gemm(out1, nullptr, 64, gb_w2, nullptr, nullptr, hw2, NB, 32, 64, 0, Abf, Wbf, stream);
  gcn_layer(hw2, deg, gb_b2, src_b, dst_b, EB, NB, 5, z_b, stream); // F=32
  k_decode<<<cdiv(EB, 256), 256, 0, stream>>>(z_b, src_b, dst_b, EB, adj_b);

  // ============ POIs: MLP stack ============
  for (int c = 0; c < NP / CHUNK; ++c) {
    long r0 = (long)c * CHUNK;
    gemm(in_pc + r0 * 401, nullptr, 401, pc_w1, pc_b1, pc_a1, hidden, CHUNK, 512, 401, 1, Abf, Wbf, stream);
    gemm(hidden, nullptr, 512, pc_w2, pc_b2, pc_a2, xb + r0 * 64, CHUNK, 64, 512, 1, Abf, Wbf, stream);
  }

  // ============ POIs: GCN encoder ============
  gemm(xb, nullptr, 64, gp_w1, nullptr, nullptr, hw, NP, 64, 64, 0, Abf, Wbf, stream);
  k_fill<<<cdiv(NP, 256), 256, 0, stream>>>(deg, 1.0f, NP);
  k_deg<<<cdiv(EP, 256), 256, 0, stream>>>(dst_p, EP, deg);
  k_rsqrt<<<cdiv(NP, 256), 256, 0, stream>>>(deg, NP);
  gcn_layer(hw, deg, gp_b1, src_p, dst_p, EP, NP, 6, out1, stream);
  k_relu<<<cdiv((long)NP * 64, 256), 256, 0, stream>>>(out1, (long)NP * 64);
  gemm(out1, nullptr, 64, gp_w2, nullptr, nullptr, hw2, NP, 32, 64, 0, Abf, Wbf, stream);
  gcn_layer(hw2, deg, gp_b2, src_p, dst_p, EP, NP, 5, z_p, stream);
  k_decode<<<cdiv(EP, 256), 256, 0, stream>>>(z_p, src_p, dst_p, EP, adj_p);
}
